// GromovWassersteinEmbedding_32762010534518
// MI455X (gfx1250) — matrix operand order repaired and checked by reading, not compiled, not run on hardware
//
#include <hip/hip_runtime.h>

// GromovWassersteinEmbedding fused pipeline for MI455X (gfx1250, wave32).
//
// d_w  = sum( (1 - exp(-(1 - (E1@E2^T)/(||E1||*||E2||^T + 1e-16)))) * trans )
// reg  = ||E1^T E1 - I||_F^2 + ||E2^T E2 - I||_F^2
//
// Strategy: convert embeddings (8 MB) to f16, run both GEMMs on
// v_wmma_f32_16x16x32_f16 (f32 accumulate), fuse the cosine/exp/trans
// epilogue so the 256 MB `trans` tensor is streamed exactly once.

typedef _Float16 half8 __attribute__((ext_vector_type(8)));
typedef _Float16 v16h  __attribute__((ext_vector_type(16)));
typedef float    v8f   __attribute__((ext_vector_type(8)));

#define GW_NUM1 8192
#define GW_NUM2 8192
#define GW_DIM  256

static __device__ __forceinline__ v16h cat16(half8 lo, half8 hi) {
    return __builtin_shufflevector(lo, hi, 0,1,2,3,4,5,6,7,8,9,10,11,12,13,14,15);
}

// A-matrix fragment (16x32 f16, M x K), X row-major with leading dim ldk.
// ISA 7.12.2 16-bit A layout: lanes 0-15 hold M=0..15 with K = {h8*8..+7, +16..+23},
// lanes 16-31 hold the complementary K groups. Two b128 loads per lane.
static __device__ __forceinline__ v16h load_fragA(const _Float16* __restrict__ X,
                                                  int ldk, int row0, int k0, int lane) {
    int r = lane & 15;
    int h = lane >> 4;
    const _Float16* p = X + (size_t)(row0 + r) * (size_t)ldk + (size_t)(k0 + h * 8);
    half8 lo = *(const half8*)(p);        // K = k0 + h*8 .. +7
    half8 hi = *(const half8*)(p + 16);   // K = k0 + h*8 + 16 .. +23
    return cat16(lo, hi);
}

// B-matrix fragment (32x16 f16, K x N) where B = Y^T, Y row-major (ldk).
// Lane n holds column n of B (= row n of Y); K matches the A layout:
// K = {k0+h*8..+7, k0+h*8+16..+23}.
static __device__ __forceinline__ v16h load_fragB(const _Float16* __restrict__ Y,
                                                  int ldk, int col0, int k0, int lane) {
    int n = lane & 15;
    int h = lane >> 4;
    const _Float16* p = Y + (size_t)(col0 + n) * (size_t)ldk + (size_t)(k0 + h * 8);
    half8 lo = *(const half8*)(p);
    half8 hi = *(const half8*)(p + 16);
    return cat16(lo, hi);
}

static __device__ __forceinline__ float wave_reduce_add(float v) {
#pragma unroll
    for (int off = 16; off > 0; off >>= 1)
        v += __shfl_xor(v, off, 32);
    return v;
}

static __device__ __forceinline__ float block_reduce_add(float v) {
    __shared__ float smem[8];
    int lane = threadIdx.x & 31;
    int w    = threadIdx.x >> 5;
    v = wave_reduce_add(v);
    __syncthreads();               // protect smem across repeated calls
    if (lane == 0) smem[w] = v;
    __syncthreads();
    float t = 0.0f;
    if (w == 0) {
        t = (lane < (int)(blockDim.x >> 5)) ? smem[lane] : 0.0f;
        t = wave_reduce_add(t);
    }
    return t;   // valid on thread 0
}

// ---------------------------------------------------------------------------
// Kernel 1: gather rows via index, compute f32 row norms, emit f16 row-major
// copy Xh[r][d] and f16 transpose Xt[d][r]. One wave per row (DIM=256 -> 8
// floats per lane).
// ---------------------------------------------------------------------------
__global__ __launch_bounds__(256)
void gw_convert_kernel(const float* __restrict__ E, const int* __restrict__ idx,
                       _Float16* __restrict__ Xh, _Float16* __restrict__ Xt,
                       float* __restrict__ energy, int nrows) {
    int w    = threadIdx.x >> 5;
    int lane = threadIdx.x & 31;
    int row  = blockIdx.x * 8 + w;
    if (row >= nrows) return;
    int src = idx[row];

    const float4* s = (const float4*)(E + (size_t)src * GW_DIM);
    float4 f0 = s[lane * 2 + 0];
    float4 f1 = s[lane * 2 + 1];

    float ss = f0.x * f0.x + f0.y * f0.y + f0.z * f0.z + f0.w * f0.w
             + f1.x * f1.x + f1.y * f1.y + f1.z * f1.z + f1.w * f1.w;

    half8 h;
    h[0] = (_Float16)f0.x; h[1] = (_Float16)f0.y;
    h[2] = (_Float16)f0.z; h[3] = (_Float16)f0.w;
    h[4] = (_Float16)f1.x; h[5] = (_Float16)f1.y;
    h[6] = (_Float16)f1.z; h[7] = (_Float16)f1.w;

    *(half8*)(Xh + (size_t)row * GW_DIM + lane * 8) = h;

    int c0 = lane * 8;
#pragma unroll
    for (int j = 0; j < 8; ++j)
        Xt[(size_t)(c0 + j) * (size_t)nrows + row] = h[j];

    ss = wave_reduce_add(ss);
    if (lane == 0) energy[row] = sqrtf(ss);
}

// ---------------------------------------------------------------------------
// Kernel 2: inner = E1 @ E2^T fused with cost epilogue and trans reduction.
// Block = 8 waves covering a 128x64 output tile; each wave: 16 rows x 64 cols
// (4 WMMA accumulators sharing one A fragment). K = 256 in steps of 32.
// ---------------------------------------------------------------------------
__global__ __launch_bounds__(256)
void gw_cost_kernel(const _Float16* __restrict__ Ah, const _Float16* __restrict__ Bh,
                    const float* __restrict__ e1, const float* __restrict__ e2,
                    const float* __restrict__ trans, float* __restrict__ partials) {
    int w    = threadIdx.x >> 5;
    int lane = threadIdx.x & 31;
    int row0 = blockIdx.y * 128 + w * 16;
    int col0 = blockIdx.x * 64;

    v8f acc0 = {}, acc1 = {}, acc2 = {}, acc3 = {};

#pragma unroll
    for (int k0 = 0; k0 < GW_DIM; k0 += 32) {
        v16h a  = load_fragA(Ah, GW_DIM, row0, k0, lane);
        v16h b0 = load_fragB(Bh, GW_DIM, col0 +  0, k0, lane);
        v16h b1 = load_fragB(Bh, GW_DIM, col0 + 16, k0, lane);
        v16h b2 = load_fragB(Bh, GW_DIM, col0 + 32, k0, lane);
        v16h b3 = load_fragB(Bh, GW_DIM, col0 + 48, k0, lane);
        acc0 = __builtin_amdgcn_wmma_f32_16x16x32_f16(false, a, false, b0, (short)0, acc0, false, false);
        acc1 = __builtin_amdgcn_wmma_f32_16x16x32_f16(false, a, false, b1, (short)0, acc1, false, false);
        acc2 = __builtin_amdgcn_wmma_f32_16x16x32_f16(false, a, false, b2, (short)0, acc2, false, false);
        acc3 = __builtin_amdgcn_wmma_f32_16x16x32_f16(false, a, false, b3, (short)0, acc3, false, false);
    }

    // C/D layout: lanes 0-15 -> M = vgpr, lanes 16-31 -> M = vgpr + 8; N = lane & 15.
    int n0 = col0 + (lane & 15);
    int mb = row0 + (lane >> 4) * 8;

    float e1v[8];
#pragma unroll
    for (int v = 0; v < 8; ++v) e1v[v] = e1[mb + v];

    v8f accs[4] = { acc0, acc1, acc2, acc3 };
    float sum = 0.0f;
#pragma unroll
    for (int t = 0; t < 4; ++t) {
        int n = n0 + t * 16;
        float e2v = e2[n];
#pragma unroll
        for (int v = 0; v < 8; ++v) {
            float inner = accs[t][v];
            float denom = __fmaf_rn(e1v[v], e2v, 1e-16f);
            float cost  = 1.0f - __expf(-(1.0f - inner / denom));
            sum = __fmaf_rn(cost, trans[(size_t)(mb + v) * GW_NUM2 + n], sum);
        }
    }

    sum = block_reduce_add(sum);
    if (threadIdx.x == 0)
        partials[blockIdx.y * gridDim.x + blockIdx.x] = sum;
}

// ---------------------------------------------------------------------------
// Kernel 3: regularizer. E^T E = Et @ Et^T with Et (256 x 8192) f16; output is
// 256x256 -> 256 tiles of 16x16 per matrix, one wave each, K = 8192 (256 WMMAs).
// Epilogue: sum over (x - delta_mn)^2.
// ---------------------------------------------------------------------------
__global__ __launch_bounds__(256)
void gw_reg_kernel(const _Float16* __restrict__ Et1, const _Float16* __restrict__ Et2,
                   float* __restrict__ partials, int nrows) {
    int w    = threadIdx.x >> 5;
    int lane = threadIdx.x & 31;
    const _Float16* Xt = (blockIdx.y == 0) ? Et1 : Et2;

    int tile = blockIdx.x * 8 + w;     // 0..255
    int row0 = (tile >> 4) * 16;
    int col0 = (tile & 15) * 16;

    v8f acc = {};
    for (int k0 = 0; k0 < nrows; k0 += 32) {
        v16h a = load_fragA(Xt, nrows, row0, k0, lane);
        v16h b = load_fragB(Xt, nrows, col0, k0, lane);
        acc = __builtin_amdgcn_wmma_f32_16x16x32_f16(false, a, false, b, (short)0, acc, false, false);
    }

    int n  = col0 + (lane & 15);
    int mb = row0 + (lane >> 4) * 8;
    float sum = 0.0f;
#pragma unroll
    for (int v = 0; v < 8; ++v) {
        float d = acc[v] - (((mb + v) == n) ? 1.0f : 0.0f);
        sum = __fmaf_rn(d, d, sum);
    }

    sum = block_reduce_add(sum);
    if (threadIdx.x == 0)
        partials[blockIdx.y * gridDim.x + blockIdx.x] = sum;
}

// ---------------------------------------------------------------------------
// Kernel 4: final reduction -> d_out[0] = d_w, d_out[1] = regularizer.
// Writes (not accumulates), so poisoned buffers are harmless and replays are
// deterministic.
// ---------------------------------------------------------------------------
__global__ __launch_bounds__(256)
void gw_reduce_final(const float* __restrict__ costP, int nCost,
                     const float* __restrict__ regP, int nReg,
                     float* __restrict__ out) {
    float s = 0.0f;
    for (int i = threadIdx.x; i < nCost; i += 256) s += costP[i];
    s = block_reduce_add(s);
    if (threadIdx.x == 0) out[0] = s;
    __syncthreads();
    float r = 0.0f;
    for (int i = threadIdx.x; i < nReg; i += 256) r += regP[i];
    r = block_reduce_add(r);
    if (threadIdx.x == 0) out[1] = r;
}

extern "C" void kernel_launch(void* const* d_in, const int* in_sizes, int n_in,
                              void* d_out, int out_size, void* d_ws, size_t ws_size,
                              hipStream_t stream) {
    (void)in_sizes; (void)n_in; (void)out_size; (void)ws_size;

    const int*   index1 = (const int*)d_in[0];
    const int*   index2 = (const int*)d_in[1];
    const float* trans  = (const float*)d_in[2];
    const float* emb1   = (const float*)d_in[3];
    const float* emb2   = (const float*)d_in[4];
    float* out = (float*)d_out;

    // Workspace carve-up (~17 MB total), 256B-aligned slices.
    char* ws = (char*)d_ws;
    size_t off = 0;
    auto carve = [&](size_t bytes) -> void* {
        void* p = ws + off;
        off += (bytes + 255) & ~(size_t)255;
        return p;
    };
    _Float16* A1h = (_Float16*)carve((size_t)GW_NUM1 * GW_DIM * sizeof(_Float16));
    _Float16* A2h = (_Float16*)carve((size_t)GW_NUM2 * GW_DIM * sizeof(_Float16));
    _Float16* Et1 = (_Float16*)carve((size_t)GW_DIM * GW_NUM1 * sizeof(_Float16));
    _Float16* Et2 = (_Float16*)carve((size_t)GW_DIM * GW_NUM2 * sizeof(_Float16));
    float* e1    = (float*)carve((size_t)GW_NUM1 * sizeof(float));
    float* e2    = (float*)carve((size_t)GW_NUM2 * sizeof(float));
    const int nCostBlocksX = GW_NUM2 / 64;    // 128
    const int nCostBlocksY = GW_NUM1 / 128;   // 64
    const int nCost = nCostBlocksX * nCostBlocksY;  // 8192
    const int nRegBlocksX = 32;               // 256 tiles / 8 waves
    const int nReg = nRegBlocksX * 2;         // 64
    float* costP = (float*)carve((size_t)nCost * sizeof(float));
    float* regP  = (float*)carve((size_t)nReg * sizeof(float));

    gw_convert_kernel<<<dim3(GW_NUM1 / 8), 256, 0, stream>>>(emb1, index1, A1h, Et1, e1, GW_NUM1);
    gw_convert_kernel<<<dim3(GW_NUM2 / 8), 256, 0, stream>>>(emb2, index2, A2h, Et2, e2, GW_NUM2);

    gw_cost_kernel<<<dim3(nCostBlocksX, nCostBlocksY), 256, 0, stream>>>(
        A1h, A2h, e1, e2, trans, costP);

    gw_reg_kernel<<<dim3(nRegBlocksX, 2), 256, 0, stream>>>(Et1, Et2, regP, GW_NUM1);

    gw_reduce_final<<<1, 256, 0, stream>>>(costP, nCost, regP, nReg, out);
}
